// DRSM_79302276153939
// MI455X (gfx1250) — matrix operand order repaired and checked by reading, not compiled, not run on hardware
//
#include <hip/hip_runtime.h>
#include <hip/hip_bf16.h>

typedef __attribute__((ext_vector_type(16))) _Float16 v16h;
typedef __attribute__((ext_vector_type(8)))  _Float16 v8h;
typedef __attribute__((ext_vector_type(8)))  float    v8f;

// Problem constants
#define BSZ   4
#define CIN   64
#define OUTC  64
#define HW    16384      // 128*128
#define HDIM  128

// Workspace layout (units: _Float16 elements). All offsets multiples of 8 (16B aligned).
#define XM_OFF  0                       // [4][192][128][128] f16  = 12,582,912
#define CAT_OFF 12582912                // [4][256][128][128] f16  = 16,777,216
#define WT_OFF  29360128                // [4][64][1728] f16       =    442,368
#define WO_OFF  29802496                // [64][2304] f16          =    147,456

#define OUT_MAIN_ELEMS 4194304         // 4*64*128*128 floats (d_out head)
// masks live at d_out + OUT_MAIN_ELEMS : 4*3*128*128 floats

// -------------------------------------------------------------------------
// Weight repack to K-order k = tap*NPLANES + plane (tap uniform per 32-chunk).
// Branch: Wt[b][o][t*192 + m*64 + c] = kernel[b][m][o][c][t]   (fp32 -> f16)
// -------------------------------------------------------------------------
__global__ void prep_wt_kernel(const float* __restrict__ kern, _Float16* __restrict__ Wt) {
    int i = blockIdx.x * 256 + threadIdx.x;
    if (i >= BSZ * OUTC * 1728) return;
    int q = i;
    int c = q % 64; q /= 64;
    int m = q % 3;  q /= 3;
    int t = q % 9;  q /= 9;
    int o = q % 64; int b = q / 64;
    int src = ((((b * 3 + m) * 64 + o) * 64 + c) * 9) + t;
    Wt[i] = (_Float16)kern[src];
}

// conv_out: Wo[o][t*256 + cin] = conv_out_w[o][cin][t]  (fp32 -> f16)
__global__ void prep_wo_kernel(const float* __restrict__ w, _Float16* __restrict__ Wo) {
    int i = blockIdx.x * 256 + threadIdx.x;
    if (i >= OUTC * 2304) return;
    int q = i;
    int cin = q % 256; q /= 256;
    int t   = q % 9;   int o = q / 9;
    int src = (o * 256 + cin) * 9 + t;
    Wo[i] = (_Float16)w[src];
}

// -------------------------------------------------------------------------
// Mask conv (3ch, 3x3) + softmax over channels + xm = x * mask (f16, 192 planes/batch)
// One block per (b,y) row, 128 threads = x columns.
// -------------------------------------------------------------------------
__global__ void mask_xm_kernel(const float* __restrict__ x,
                               const float* __restrict__ mw,   // [3][64][3][3]
                               const float* __restrict__ mb,   // [3]
                               float* __restrict__ maskOut,    // [4][3][128][128]
                               _Float16* __restrict__ xm) {    // [4][192][128][128]
    __shared__ float wl[1728];
    for (int i = threadIdx.x; i < 1728; i += 128) wl[i] = mw[i];
    __syncthreads();

    const int b = blockIdx.x >> 7;
    const int y = blockIdx.x & 127;
    const int xc = threadIdx.x;

    float s0 = mb[0], s1 = mb[1], s2 = mb[2];
    const float* xb = x + (size_t)b * CIN * HW;
    for (int c = 0; c < CIN; ++c) {
        const float* xch = xb + c * HW;
        const float* wc0 = wl + c * 9;
        #pragma unroll
        for (int ky = 0; ky < 3; ++ky) {
            int iy = y + ky - 1;
            if ((unsigned)iy >= 128u) continue;
            const float* row = xch + iy * HDIM;
            #pragma unroll
            for (int kx = 0; kx < 3; ++kx) {
                int ix = xc + kx - 1;
                if ((unsigned)ix >= 128u) continue;
                float xv = row[ix];
                int t = ky * 3 + kx;
                s0 += xv * wc0[t];
                s1 += xv * wc0[576 + t];
                s2 += xv * wc0[1152 + t];
            }
        }
    }
    float mmax = fmaxf(s0, fmaxf(s1, s2));
    float e0 = __expf(s0 - mmax), e1 = __expf(s1 - mmax), e2 = __expf(s2 - mmax);
    float rs = 1.0f / (e0 + e1 + e2);
    float m0 = e0 * rs, m1 = e1 * rs, m2 = e2 * rs;

    size_t mo = (size_t)(b * 3) * HW + y * HDIM + xc;
    maskOut[mo]          = m0;
    maskOut[mo + HW]     = m1;
    maskOut[mo + 2 * HW] = m2;

    _Float16* xmb = xm + (size_t)b * 192 * HW + y * HDIM + xc;
    const float* xpix = xb + y * HDIM + xc;
    for (int c = 0; c < CIN; ++c) {
        float xv = xpix[c * HW];
        xmb[(size_t)c * HW]         = (_Float16)(xv * m0);
        xmb[(size_t)(64 + c) * HW]  = (_Float16)(xv * m1);
        xmb[(size_t)(128 + c) * HW] = (_Float16)(xv * m2);
    }
}

// -------------------------------------------------------------------------
// WMMA fragment loads from LDS, contiguous k order (k = 16*hi + s), same in A and B.
// A (16x32 f16): lane M=l%16; halves K = [hi*16, hi*16+16).
// B (32x16 f16): lane N=l%16; halves K = [hi*16, hi*16+16).
// LDS row stride 40 halves (80B): 16B alignment + bank spread.
// -------------------------------------------------------------------------
__device__ __forceinline__ v16h load_fragA(const _Float16* base, int row, int hi) {
    union { v16h v; v8h h[2]; } u;
    u.h[0] = *(const v8h*)(base + row * 40 + hi * 16);
    u.h[1] = *(const v8h*)(base + row * 40 + hi * 16 + 8);
    return u.v;
}
__device__ __forceinline__ v16h load_fragB(const _Float16* base, int n, int hi) {
    union { v16h v; v8h h[2]; } u;
    u.h[0] = *(const v8h*)(base + n * 40 + hi * 16);
    u.h[1] = *(const v8h*)(base + n * 40 + hi * 16 + 8);
    return u.v;
}

// -------------------------------------------------------------------------
// Implicit-GEMM dilated 3x3 conv, f16 in / f32 accumulate via WMMA.
// Block = 128 threads (4 waves). One block computes one output row:
//   M = 64 out channels, N = 128 pixels (x), K = NPLANES*9 (k = tap*NPLANES+p).
// Double-buffered LDS + register staging: ONE barrier per k-step.
//   iter kk: global loads (chunk kk+1) -> compute buf[kk&1] (8 WMMAs/wave)
//            -> ds_store staged regs into buf[1-(kk&1)] -> barrier.
// __launch_bounds__(128, 1): no scratch spills (latency hidden by pipeline).
// MODE 0: branch conv -> f16 store into cat at channel base dil_idx*64.
// MODE 1: conv_out    -> bias + BN(eval) + ReLU, f32 store to d_out.
// -------------------------------------------------------------------------
template <int NPLANES, int MODE>
__global__ __launch_bounds__(128, 1)
void igemm_conv_kernel(const _Float16* __restrict__ xin,
                       const _Float16* __restrict__ wbase,
                       _Float16* __restrict__ catOut,
                       float* __restrict__ outF,
                       const float* __restrict__ bias,
                       const float* __restrict__ bg,
                       const float* __restrict__ bb,
                       const float* __restrict__ bm,
                       const float* __restrict__ bv) {
    constexpr int K   = NPLANES * 9;
    constexpr int NK  = K / 32;
    constexpr int CPT = NPLANES / 32;   // K-chunks per tap (6 or 8)

    __shared__ __align__(16) _Float16 AldsBuf[2][64 * 40];
    __shared__ __align__(16) _Float16 BldsBuf[2][128 * 40];

    const int y = blockIdx.x;
    int b, dil, chBase;
    if constexpr (MODE == 0) {
        int combo = blockIdx.y;      // b*4 + dilation index
        b = combo >> 2;
        int dd = combo & 3;
        dil = 1 + 2 * dd;            // 1,3,5,7
        chBase = dd * 64;
    } else {
        b = blockIdx.y;
        dil = 1;
        chBase = 0;
    }
    const _Float16* inB = xin + (size_t)b * NPLANES * HW;
    const _Float16* Wg  = (MODE == 0) ? (wbase + (size_t)b * OUTC * K) : wbase;

    const int tid  = threadIdx.x;
    const int lane = tid & 31;
    const int wv   = tid >> 5;       // wave id 0..3
    const int lr   = lane & 15;
    const int hi   = lane >> 4;

    const v8f vzero = {0.f, 0.f, 0.f, 0.f, 0.f, 0.f, 0.f, 0.f};
    v8f acc[4][2];
    #pragma unroll
    for (int mt = 0; mt < 4; ++mt)
        #pragma unroll
        for (int j = 0; j < 2; ++j) acc[mt][j] = vzero;

    const int wrow = tid >> 1;       // weight row 0..63
    const int wseg = tid & 1;        // which 16-half segment
    const _Float16 hzero = (_Float16)0.f;

    // staging registers for the software pipeline
    v8h aR0, aR1, bR[4];

    auto stage_load = [&](int kk) {
        const int k0 = kk * 32;
        // A: 16 contiguous halves of this thread's weight row segment
        {
            const _Float16* srcW = Wg + (size_t)wrow * K + k0 + wseg * 16;
            aR0 = *(const v8h*)(srcW);
            aR1 = *(const v8h*)(srcW + 8);
        }
        // B: 32 consecutive planes at uniform tap; thread = pixel column.
        const int tap = kk / CPT;               // 0..8, uniform per chunk
        const int p0  = (kk - tap * CPT) * 32;  // first plane of chunk
        const int ky  = tap / 3;
        const int kx  = tap - ky * 3;
        const int iy  = y + dil * (ky - 1);
        const int ix  = tid + dil * (kx - 1);
        const bool oky = (unsigned)iy < 128u;
        const bool okx = (unsigned)ix < 128u;
        if (!oky) {
            #pragma unroll
            for (int g = 0; g < 4; ++g)
                #pragma unroll
                for (int e = 0; e < 8; ++e) bR[g][e] = hzero;
        } else {
            const int ixc = okx ? ix : 0;       // clamp address, zero result below
            const _Float16* src = inB + (size_t)p0 * HW + iy * HDIM + ixc;
            #pragma unroll
            for (int g = 0; g < 4; ++g) {
                #pragma unroll
                for (int e = 0; e < 8; ++e) {
                    _Float16 v = src[(size_t)(g * 8 + e) * HW];  // immediate offsets
                    bR[g][e] = okx ? v : hzero;
                }
            }
        }
    };

    auto stage_store = [&](int buf) {
        _Float16* Ad = AldsBuf[buf];
        _Float16* Bd = BldsBuf[buf];
        *(v8h*)(&Ad[wrow * 40 + wseg * 16])     = aR0;
        *(v8h*)(&Ad[wrow * 40 + wseg * 16 + 8]) = aR1;
        #pragma unroll
        for (int g = 0; g < 4; ++g)
            *(v8h*)(&Bd[tid * 40 + g * 8]) = bR[g];
    };

    // prologue: chunk 0 into buffer 0
    stage_load(0);
    stage_store(0);
    __syncthreads();

    for (int kk = 0; kk < NK; ++kk) {
        const int cur = kk & 1;
        if (kk + 1 < NK) stage_load(kk + 1);    // global -> regs, overlaps WMMAs

        const _Float16* Ac = AldsBuf[cur];
        const _Float16* Bc = BldsBuf[cur];
        // j outer / mt inner; compiler CSEs the A-fragment LDS loads across j.
        #pragma unroll
        for (int j = 0; j < 2; ++j) {
            v16h bf = load_fragB(Bc, wv * 32 + j * 16 + lr, hi);
            #pragma unroll
            for (int mt = 0; mt < 4; ++mt) {
                v16h af = load_fragA(Ac, mt * 16 + lr, hi);
                acc[mt][j] = __builtin_amdgcn_wmma_f32_16x16x32_f16(
                    false, af, false, bf, (short)0, acc[mt][j], false, false);
            }
        }

        if (kk + 1 < NK) stage_store(cur ^ 1);  // fill the other buffer
        __syncthreads();                        // single barrier per k-step
    }

    // ---- epilogue. D layout: N = lane%16 (pixel), M = v + 8*hi (out channel).
    if constexpr (MODE == 0) {
        size_t base = (size_t)(b * 256 + chBase) * HW + (size_t)y * HDIM;
        #pragma unroll
        for (int mt = 0; mt < 4; ++mt) {
            #pragma unroll
            for (int j = 0; j < 2; ++j) {
                int xcol = wv * 32 + j * 16 + lr;
                #pragma unroll
                for (int v = 0; v < 8; ++v) {
                    int oc = mt * 16 + hi * 8 + v;
                    catOut[base + (size_t)oc * HW + xcol] = (_Float16)acc[mt][j][v];
                }
            }
        }
    } else {
        size_t base = (size_t)b * OUTC * HW + (size_t)y * HDIM;
        #pragma unroll
        for (int mt = 0; mt < 4; ++mt) {
            #pragma unroll
            for (int v = 0; v < 8; ++v) {
                int oc = mt * 16 + hi * 8 + v;
                float inv = bg[oc] * rsqrtf(bv[oc] + 1e-5f);
                float sh  = bb[oc] - bm[oc] * inv;
                float bs  = bias[oc];
                #pragma unroll
                for (int j = 0; j < 2; ++j) {
                    int xcol = wv * 32 + j * 16 + lr;
                    float r = (acc[mt][j][v] + bs) * inv + sh;
                    r = fmaxf(r, 0.0f);
                    outF[base + (size_t)oc * HW + xcol] = r;
                }
            }
        }
    }
}

// -------------------------------------------------------------------------
extern "C" void kernel_launch(void* const* d_in, const int* in_sizes, int n_in,
                              void* d_out, int out_size, void* d_ws, size_t ws_size,
                              hipStream_t stream) {
    const float* x    = (const float*)d_in[0];
    const float* kern = (const float*)d_in[1];
    const float* mw   = (const float*)d_in[2];
    const float* mb   = (const float*)d_in[3];
    const float* wo   = (const float*)d_in[4];
    const float* ob   = (const float*)d_in[5];
    const float* bg   = (const float*)d_in[6];
    const float* bb   = (const float*)d_in[7];
    const float* bm   = (const float*)d_in[8];
    const float* bv   = (const float*)d_in[9];
    float* out = (float*)d_out;

    _Float16* ws  = (_Float16*)d_ws;
    _Float16* xm  = ws + XM_OFF;
    _Float16* cat = ws + CAT_OFF;
    _Float16* Wt  = ws + WT_OFF;
    _Float16* Wo  = ws + WO_OFF;

    // 1) repack weights to f16 GEMM layout (k = tap*NPLANES + plane)
    prep_wt_kernel<<<(BSZ * OUTC * 1728 + 255) / 256, 256, 0, stream>>>(kern, Wt);
    prep_wo_kernel<<<(OUTC * 2304 + 255) / 256, 256, 0, stream>>>(wo, Wo);

    // 2) masks (softmax) + masked f16 input planes
    mask_xm_kernel<<<BSZ * HDIM, 128, 0, stream>>>(x, mw, mb, out + OUT_MAIN_ELEMS, xm);

    // 3) four dilated grouped branches (mask-sum folded into K) -> cat (f16)
    igemm_conv_kernel<192, 0><<<dim3(HDIM, BSZ * 4), 128, 0, stream>>>(
        xm, Wt, cat, nullptr, nullptr, nullptr, nullptr, nullptr, nullptr);

    // 4) conv_out + bias + BN(eval) + ReLU -> d_out (f32)
    igemm_conv_kernel<256, 1><<<dim3(HDIM, BSZ), 128, 0, stream>>>(
        cat, Wo, nullptr, out, ob, bg, bb, bm, bv);
}
